// OOQKVAttentionG2_18021682775016
// MI455X (gfx1250) — matrix operand order repaired and checked by reading, not compiled, not run on hardware
//
#include <hip/hip_runtime.h>
#include <hip/hip_bf16.h>
#include <stdint.h>

// ---------------------------------------------------------------------------
// Types for CDNA5 WMMA (wave32): v16bf A/B fragments, v8f f32 accumulator.
// ---------------------------------------------------------------------------
typedef __bf16 bf16;
typedef __attribute__((ext_vector_type(8)))  bf16  bf16x8;
typedef __attribute__((ext_vector_type(16))) bf16  v16bf;
typedef __attribute__((ext_vector_type(8)))  float v8f;

typedef unsigned int u32;
typedef __attribute__((ext_vector_type(4))) u32 u32x4;
typedef __attribute__((ext_vector_type(4))) int i32x4;
typedef __attribute__((ext_vector_type(8))) int i32x8;

// ---- Tensor Data Mover availability (probe-verified builtin on both lanes) --
#ifndef __HIP_DEVICE_COMPILE__
#define HAVE_TDM 0
#elif __has_builtin(__builtin_amdgcn_tensor_load_to_lds)
#define HAVE_TDM 1
#else
#define HAVE_TDM 0
#endif

__device__ __forceinline__ v8f wmma_bf16(v16bf a, v16bf b, v8f c) {
    // D = A(16x32 bf16) * B(32x16 bf16) + C(16x16 f32)
    return __builtin_amdgcn_wmma_f32_16x16x32_bf16(
        false, a, false, b, (short)0, c, false, false);
}

// A fragment (16x32 bf16), row-major source, K contiguous per row.
// lane l: row = l&15 ; K chunks at koff=(l>>4)*8 and koff+16.
__device__ __forceinline__ v16bf load_frag_a(const bf16* base, int stride) {
    int l = threadIdx.x & 31;
    const bf16* p = base + (l & 15) * stride + ((l >> 4) << 3);
    bf16x8 lo = *(const bf16x8*)p;
    bf16x8 hi = *(const bf16x8*)(p + 16);
    return __builtin_shufflevector(lo, hi, 0,1,2,3,4,5,6,7,8,9,10,11,12,13,14,15);
}

// B fragment (32x16 bf16) from memory where each column's K elems are
// contiguous. lane l: col = l&15 ; K chunks at koff=(l>>4)*8 and koff+16.
__device__ __forceinline__ v16bf load_frag_b(const bf16* base, int stride) {
    int l = threadIdx.x & 31;
    const bf16* p = base + (l & 15) * stride + ((l >> 4) << 3);
    bf16x8 lo = *(const bf16x8*)p;
    bf16x8 hi = *(const bf16x8*)(p + 16);
    return __builtin_shufflevector(lo, hi, 0,1,2,3,4,5,6,7,8,9,10,11,12,13,14,15);
}

// ---------------------------------------------------------------------------
// TDM: async 2-D tile load Global -> LDS.  D# per CDNA5 ISA
// group0: [1:0]=count=1 | lds_addr | global_addr[56:0] | type=2
// group1: data_size=1(2B) | tensor_dim0/1 | tile_dim0/1 | dim0_stride
// ---------------------------------------------------------------------------
#if HAVE_TDM
__device__ __forceinline__ void tdm_load_2d(u32 lds_byte_off, const void* gsrc,
                                            u32 tile_x, u32 tile_rows, u32 stride_el) {
    unsigned long long ga = (unsigned long long)(uintptr_t)gsrc;
    u32x4 g0;
    g0.x = 1u;                                            // count=1, user desc
    g0.y = lds_byte_off;                                  // LDS byte address
    g0.z = (u32)ga;                                       // global addr [31:0]
    g0.w = (u32)((ga >> 32) & 0x01ffffffu) | (2u << 30);  // addr[56:32] | type=2
    u32 td0 = stride_el;          // row length for OOB (>= tile_x, never trips)
    u32 td1 = 1u << 20;           // plenty of rows, never trips OOB
    i32x8 g1;
    g1[0] = (int)(1u << 16);                              // data_size=1 -> 2 bytes
    g1[1] = (int)((td0 & 0xffffu) << 16);                 // tensor_dim0[15:0]
    g1[2] = (int)((td0 >> 16) | ((td1 & 0xffffu) << 16)); // td0[31:16]|td1[15:0]
    g1[3] = (int)((td1 >> 16) | (tile_x << 16));          // td1[31:16]|tile_dim0
    g1[4] = (int)(tile_rows & 0xffffu);                   // tile_dim1 (tile_dim2=0)
    g1[5] = (int)stride_el;                               // tensor_dim0_stride[31:0]
    g1[6] = 0;
    g1[7] = 0;
    i32x4 z4 = {0, 0, 0, 0};
#if defined(__clang_major__) && (__clang_major__ >= 23)
    i32x8 z8 = {0, 0, 0, 0, 0, 0, 0, 0};
    __builtin_amdgcn_tensor_load_to_lds(g0, g1, z4, z4, z8, 0);
#else
    __builtin_amdgcn_tensor_load_to_lds(g0, g1, z4, z4, 0);
#endif
}

__device__ __forceinline__ void wait_tensor0() {
#if __has_builtin(__builtin_amdgcn_s_wait_tensorcnt)
    __builtin_amdgcn_s_wait_tensorcnt(0);
#else
    asm volatile("s_wait_tensorcnt 0" ::: "memory");
#endif
}

// Low 32 bits of a generic pointer to __shared__ == LDS byte offset (ISA 10.2).
__device__ __forceinline__ u32 lds_off(const void* p) {
    return (u32)(uintptr_t)p;
}
#else
// Fallback: cooperative synchronous tile copy (whole block participates).
__device__ __forceinline__ void coop_copy_tile(bf16* dst, const bf16* src,
                                               int rows, int elems, int stride) {
    int chunks_per_row = elems >> 3;                    // 8 bf16 per chunk
    int total = rows * chunks_per_row;
    for (int i = threadIdx.x; i < total; i += blockDim.x) {
        int r = i / chunks_per_row, c = (i - r * chunks_per_row) << 3;
        *(bf16x8*)&dst[r * elems + c] = *(const bf16x8*)&src[(long)r * stride + c];
    }
}
#endif

// ---------------------------------------------------------------------------
// Prep kernels: fp32 -> bf16 convert; tiled transpose-convert for weights so
// GEMM B fragments are K-contiguous (out[o, i] = bf16(in[i, o])).
// ---------------------------------------------------------------------------
__global__ void cvt_bf16_kernel(const float* __restrict__ in, bf16* __restrict__ out, int n) {
    int i = blockIdx.x * blockDim.x + threadIdx.x;
    if (i < n) out[i] = (bf16)in[i];
}

__global__ void transpose_cvt_kernel(const float* __restrict__ in, bf16* __restrict__ out,
                                     int I, int O) {
    __shared__ float tile[32][33];
    int o0 = blockIdx.x * 32, i0 = blockIdx.y * 32;
    int tx = threadIdx.x & 31, ty = threadIdx.x >> 5;
    for (int r = ty; r < 32; r += 8)
        tile[r][tx] = in[(long)(i0 + r) * O + o0 + tx];
    __syncthreads();
    for (int r = ty; r < 32; r += 8)
        out[(long)(o0 + r) * I + i0 + tx] = (bf16)tile[tx][r];
}

// ---------------------------------------------------------------------------
// Kernel 1: QKV projection GEMM.  y = x @ W + b, M=4096, N=512, K=512.
// grid (128, 8, 3), block 256 (8 waves = 2 rowblocks x 4 col-quarters).
// ---------------------------------------------------------------------------
__global__ void proj_kernel(const bf16* __restrict__ xb,
                            const bf16* __restrict__ WqT, const bf16* __restrict__ WkT,
                            const bf16* __restrict__ WvT,
                            const float* __restrict__ bq, const float* __restrict__ bk,
                            const float* __restrict__ bv,
                            float* __restrict__ qf, bf16* __restrict__ kb,
                            bf16* __restrict__ vt) {
    int which = blockIdx.z;
    const bf16*  WT   = (which == 0) ? WqT : (which == 1) ? WkT : WvT;
    const float* bias = (which == 0) ? bq  : (which == 1) ? bk  : bv;
    int wave = threadIdx.x >> 5, lane = threadIdx.x & 31;
    int rb = wave >> 2, cq = wave & 3;
    int row0 = blockIdx.x * 32 + rb * 16;
    int col0 = blockIdx.y * 64 + cq * 16;
    int hl = lane >> 4;

    v8f acc0 = {}, acc1 = {};                      // two chains for ILP
#pragma unroll
    for (int ks = 0; ks < 16; ks += 2) {
        v16bf a0 = load_frag_a(xb + row0 * 512 + ks * 32, 512);
        v16bf b0 = load_frag_b(WT + (long)col0 * 512 + ks * 32, 512);
        acc0 = wmma_bf16(a0, b0, acc0);
        v16bf a1 = load_frag_a(xb + row0 * 512 + (ks + 1) * 32, 512);
        v16bf b1 = load_frag_b(WT + (long)col0 * 512 + (ks + 1) * 32, 512);
        acc1 = wmma_bf16(a1, b1, acc1);
    }
    int coll = col0 + (lane & 15);
    float bsv = bias[coll];
#pragma unroll
    for (int r = 0; r < 8; ++r) {
        int t = row0 + r + 8 * hl;
        float val = acc0[r] + acc1[r] + bsv;
        if (which == 0) {
            qf[t * 512 + coll] = val;
        } else if (which == 1) {
            kb[t * 512 + coll] = (bf16)val;
        } else {
            int bidx = t >> 10, n = t & 1023;
            int h = coll >> 6, d = coll & 63;
            vt[(((bidx << 3) + h) * 64 + d) * 1024 + n] = (bf16)val;
        }
    }
}

// ---------------------------------------------------------------------------
// Kernel 2 (dominant, 137 GFLOP): fused
//   qg[t,h,e] = sum_d q[t,h,d] * gelu((x@Wg+bg)[t,h,d,e])   (g never stored)
// grid (128, 8 heads), block 256 (8 waves = 2 rowblocks x 4 e-quarters).
// A fragments of x pinned in 128 VGPRs. B tiles (64 cols x 256 K, 32 KB)
// streamed by the Tensor Data Mover into a double-buffered LDS tile shared by
// all 8 waves; TDM of phase p+1 overlaps the 8 WMMAs of phase p.
// ---------------------------------------------------------------------------
__global__ void qg_kernel(const bf16* __restrict__ xb, const bf16* __restrict__ WgT,
                          const float* __restrict__ bg, const float* __restrict__ qf,
                          bf16* __restrict__ qg) {
    __shared__ __align__(16) bf16 btile[2][64 * 256];     // 2 x 32 KB
    int wave = threadIdx.x >> 5, lane = threadIdx.x & 31;
    int rb = wave >> 2, eq = wave & 3;
    int h = blockIdx.y;
    int row0 = blockIdx.x * 32 + rb * 16;
    int hl = lane >> 4;

    v16bf afr[16];
#pragma unroll
    for (int ks = 0; ks < 16; ++ks)
        afr[ks] = load_frag_a(xb + row0 * 512 + ks * 32, 512);

    float qacc[8];
#pragma unroll
    for (int r = 0; r < 8; ++r) qacc[r] = 0.0f;

    const bf16* wg_h = WgT + (long)h * 4096 * 512;        // head's 4096 columns

#if HAVE_TDM
    if (wave == 0) {
        tdm_load_2d(lds_off(&btile[0][0]), wg_h, 256, 64, 512);  // phase 0
        wait_tensor0();
    }
    __syncthreads();
#endif

    for (int d = 0; d < 64; ++d) {
        v8f g0 = {}, g1 = {};
#pragma unroll 1
        for (int kp = 0; kp < 2; ++kp) {
            int p = d * 2 + kp;
            int buf = p & 1;
#if HAVE_TDM
            if (wave == 0 && p + 1 < 128) {
                int pn = p + 1, dn = pn >> 1, kpn = pn & 1;
                const bf16* src = wg_h + ((long)dn * 64) * 512 + kpn * 256;
                tdm_load_2d(lds_off(&btile[buf ^ 1][0]), src, 256, 64, 512);
            }
#else
            coop_copy_tile(&btile[buf][0],
                           wg_h + ((long)d * 64) * 512 + kp * 256, 64, 256, 512);
            __syncthreads();
#endif
            const bf16* bt = &btile[buf][0] + (eq * 16) * 256;  // wave's 16 cols
#pragma unroll
            for (int k2 = 0; k2 < 4; ++k2) {
                int ks = kp * 8 + k2 * 2;
                v16bf bf0 = load_frag_b(bt + (k2 * 2) * 32, 256);
                v16bf bf1 = load_frag_b(bt + (k2 * 2 + 1) * 32, 256);
                g0 = wmma_bf16(afr[ks], bf0, g0);
                g1 = wmma_bf16(afr[ks + 1], bf1, g1);
            }
#if HAVE_TDM
            if (wave == 0) wait_tensor0();
            __syncthreads();
#else
            __syncthreads();
#endif
        }
        long cbase = (long)h * 4096 + d * 64 + eq * 16;
        float bias = bg[cbase + (lane & 15)];
#pragma unroll
        for (int r = 0; r < 8; ++r) {
            float xg  = g0[r] + g1[r] + bias;
            float gel = 0.5f * xg * (1.0f + erff(xg * 0.70710678118654752f)); // exact GELU
            float qv  = qf[(row0 + r + 8 * hl) * 512 + h * 64 + d];
            qacc[r] += qv * gel;
        }
    }
#pragma unroll
    for (int r = 0; r < 8; ++r) {
        int t = row0 + r + 8 * hl;
        qg[t * 512 + h * 64 + eq * 16 + (lane & 15)] = (bf16)qacc[r];
    }
}

// ---------------------------------------------------------------------------
// Kernel 3: flash attention (no 1/sqrt(d) scaling, per reference).
// grid (B*H=32, N/64=16), block 128 (4 waves, one 16-query tile each).
// K/V 32-key tiles (4 KB each) TDM-streamed into double-buffered LDS shared by
// all 4 waves (removes x4 redundant VMEM); P relayout via per-wave LDS tile.
// ---------------------------------------------------------------------------
__global__ void attn_kernel(const bf16* __restrict__ qg, const bf16* __restrict__ kb,
                            const bf16* __restrict__ vt, float* __restrict__ out) {
    __shared__ __align__(16) bf16 ktile[2][32 * 64];      // [key][e]  4 KB x2
    __shared__ __align__(16) bf16 vtile[2][64 * 32];      // [d][key]  4 KB x2
    __shared__ __align__(16) bf16 plds[4][16 * 32];
    int wave = threadIdx.x >> 5, lane = threadIdx.x & 31;
    int bh = blockIdx.x;
    int b = bh >> 3, h = bh & 7;
    int n0 = blockIdx.y * 64 + wave * 16;
    int t0 = b * 1024 + n0;
    int hl = lane >> 4;

    v16bf aq0 = load_frag_a(qg + t0 * 512 + h * 64, 512);
    v16bf aq1 = load_frag_a(qg + t0 * 512 + h * 64 + 32, 512);

    float m[8], ssum[8];
#pragma unroll
    for (int r = 0; r < 8; ++r) { m[r] = -1e30f; ssum[r] = 0.0f; }
    v8f o0 = {}, o1 = {}, o2 = {}, o3 = {};

    const bf16* kbase = kb + (long)b * 1024 * 512 + h * 64;
    const bf16* vbase = vt + (long)(b * 8 + h) * 64 * 1024;
    bf16* pl = plds[wave];

#if HAVE_TDM
    if (wave == 0) {
        tdm_load_2d(lds_off(&ktile[0][0]), kbase, 64, 32, 512);
        tdm_load_2d(lds_off(&vtile[0][0]), vbase, 32, 64, 1024);
        wait_tensor0();
    }
    __syncthreads();
#endif

    for (int c = 0; c < 32; ++c) {                        // 32-key chunks
        int buf = c & 1;
        int key0 = c * 32;
#if HAVE_TDM
        if (wave == 0 && c + 1 < 32) {
            tdm_load_2d(lds_off(&ktile[buf ^ 1][0]), kbase + (long)(key0 + 32) * 512,
                        64, 32, 512);
            tdm_load_2d(lds_off(&vtile[buf ^ 1][0]), vbase + key0 + 32,
                        32, 64, 1024);
        }
#else
        coop_copy_tile(&ktile[buf][0], kbase + (long)key0 * 512, 32, 64, 512);
        coop_copy_tile(&vtile[buf][0], vbase + key0, 64, 32, 1024);
        __syncthreads();
#endif
        const bf16* kt = &ktile[buf][0];
        const bf16* vtl = &vtile[buf][0];

        // S tile 16x32: two 16-col subtiles, two K-steps each (K = 64).
        v8f s0 = {};
        s0 = wmma_bf16(aq0, load_frag_b(kt, 64), s0);
        s0 = wmma_bf16(aq1, load_frag_b(kt + 32, 64), s0);
        v8f s1 = {};
        s1 = wmma_bf16(aq0, load_frag_b(kt + 16 * 64, 64), s1);
        s1 = wmma_bf16(aq1, load_frag_b(kt + 16 * 64 + 32, 64), s1);

        // Online softmax: row r lives across one 16-lane half -> shfl_xor tree.
        float scale[8];
#pragma unroll
        for (int r = 0; r < 8; ++r) {
            float mx = fmaxf(s0[r], s1[r]);
#pragma unroll
            for (int off = 8; off; off >>= 1) mx = fmaxf(mx, __shfl_xor(mx, off, 32));
            float mn = fmaxf(m[r], mx);
            float sc = __expf(m[r] - mn);
            float p0 = __expf(s0[r] - mn);
            float p1 = __expf(s1[r] - mn);
            float rs = p0 + p1;
#pragma unroll
            for (int off = 8; off; off >>= 1) rs += __shfl_xor(rs, off, 32);
            ssum[r] = ssum[r] * sc + rs;
            m[r] = mn; scale[r] = sc;
            s0[r] = p0; s1[r] = p1;
        }
#pragma unroll
        for (int r = 0; r < 8; ++r) {
            o0[r] *= scale[r]; o1[r] *= scale[r];
            o2[r] *= scale[r]; o3[r] *= scale[r];
        }

        // C-layout P -> per-wave LDS [16 x 32] -> reload as A fragment.
#pragma unroll
        for (int r = 0; r < 8; ++r) {
            int row = r + 8 * hl;
            pl[row * 32 +      (lane & 15)] = (bf16)s0[r];
            pl[row * 32 + 16 + (lane & 15)] = (bf16)s1[r];
        }
        asm volatile("s_wait_dscnt 0\n\ts_wait_storecnt 0" ::: "memory");
        __builtin_amdgcn_wave_barrier();
        v16bf pa = load_frag_a(pl, 32);

        // O += P @ V for four 16-wide d subtiles (V tile: keys contiguous).
        o0 = wmma_bf16(pa, load_frag_b(vtl + 0 * 16 * 32, 32), o0);
        o1 = wmma_bf16(pa, load_frag_b(vtl + 1 * 16 * 32, 32), o1);
        o2 = wmma_bf16(pa, load_frag_b(vtl + 2 * 16 * 32, 32), o2);
        o3 = wmma_bf16(pa, load_frag_b(vtl + 3 * 16 * 32, 32), o3);

#if HAVE_TDM
        if (wave == 0) wait_tensor0();
        __syncthreads();
#else
        __syncthreads();
#endif
    }

#pragma unroll
    for (int r = 0; r < 8; ++r) {
        int t = t0 + r + 8 * hl;
        float inv = 1.0f / ssum[r];
        int col = lane & 15;
        float* op = out + t * 512 + h * 64;
        op[col]      = o0[r] * inv;
        op[16 + col] = o1[r] * inv;
        op[32 + col] = o2[r] * inv;
        op[48 + col] = o3[r] * inv;
    }
}

// ---------------------------------------------------------------------------
// Host-side launch. Inputs: x, Wq, bq, Wk, bk, Wv, bv, Wg, bg (fp32).
// ---------------------------------------------------------------------------
extern "C" void kernel_launch(void* const* d_in, const int* in_sizes, int n_in,
                              void* d_out, int out_size, void* d_ws, size_t ws_size,
                              hipStream_t stream) {
    const float* x  = (const float*)d_in[0];
    const float* Wq = (const float*)d_in[1];
    const float* bq = (const float*)d_in[2];
    const float* Wk = (const float*)d_in[3];
    const float* bk = (const float*)d_in[4];
    const float* Wv = (const float*)d_in[5];
    const float* bv = (const float*)d_in[6];
    const float* Wg = (const float*)d_in[7];
    const float* bg = (const float*)d_in[8];

    const int T = 4096;          // B*N tokens
    const int E = 512;
    const int GC = 8 * 64 * 64;  // 32768 generator columns

    char* ws = (char*)d_ws;
    bf16* xb  = (bf16*)ws;  ws += (size_t)T * E * 2;        //  4 MB
    bf16* WqT = (bf16*)ws;  ws += (size_t)E * E * 2;        // 0.5 MB
    bf16* WkT = (bf16*)ws;  ws += (size_t)E * E * 2;
    bf16* WvT = (bf16*)ws;  ws += (size_t)E * E * 2;
    bf16* WgT = (bf16*)ws;  ws += (size_t)GC * E * 2;       // 32 MB (fits L2)
    float* qf = (float*)ws; ws += (size_t)T * E * 4;        //  8 MB
    bf16* kb  = (bf16*)ws;  ws += (size_t)T * E * 2;        //  4 MB
    bf16* vt  = (bf16*)ws;  ws += (size_t)T * E * 2;        //  4 MB
    bf16* qgb = (bf16*)ws;  ws += (size_t)T * E * 2;        //  4 MB

    // Stage 0: precision conversion + weight transposes (K-contiguous B).
    cvt_bf16_kernel<<<(T * E + 255) / 256, 256, 0, stream>>>(x, xb, T * E);
    transpose_cvt_kernel<<<dim3(E / 32, E / 32), 256, 0, stream>>>(Wq, WqT, E, E);
    transpose_cvt_kernel<<<dim3(E / 32, E / 32), 256, 0, stream>>>(Wk, WkT, E, E);
    transpose_cvt_kernel<<<dim3(E / 32, E / 32), 256, 0, stream>>>(Wv, WvT, E, E);
    transpose_cvt_kernel<<<dim3(GC / 32, E / 32), 256, 0, stream>>>(Wg, WgT, E, GC);

    // Stage 1: QKV projections (WMMA bf16).
    proj_kernel<<<dim3(T / 32, E / 64, 3), 256, 0, stream>>>(
        xb, WqT, WkT, WvT, bq, bk, bv, qf, kb, vt);

    // Stage 2: fused generator GEMM + gelu + q-contraction (TDM-fed).
    qg_kernel<<<dim3(T / 32, 8), 256, 0, stream>>>(xb, WgT, bg, qf, qgb);

    // Stage 3: flash attention (TDM-fed K/V tiles).
    attn_kernel<<<dim3(32, 16), 128, 0, stream>>>(qgb, kb, vt, (float*)d_out);
}
